// ResidualVQLayer_22471268892877
// MI455X (gfx1250) — matrix operand order repaired and checked by reading, not compiled, not run on hardware
//
#include <hip/hip_runtime.h>
#include <math.h>

// ---------------------------------------------------------------------------
// ResidualVQ fused pipeline for MI455X (gfx1250, wave32, WMMA bf16 16x16x32)
//   k_cvt/k_norms : pre-convert weights+codebooks to bf16, norms, zero loss
//   k_proj_vq     : x @ Wp + b (WMMA) -> 2-head VQ (WMMA distance GEMM)
//   k_ref0        : MLP(c) (WMMA) -> VQ vs cb_s0 (WMMA)
//   k_ref1        : MLP([c,e0]) (WMMA) -> VQ vs cb_s1 (WMMA)
//   k_out         : [c,e0,e1] @ out_w.T + out_b (WMMA) -> z_q
// Intermediates (c/e0/e1, bf16) + bf16 weights live in workspace; they are
// L2-resident (192 MB L2) so HBM traffic stays ~ x + z_q = 268 MB -> the
// kernel is HBM-bound at ~11.5 us; all GEMMs ride v_wmma_f32_16x16x32_bf16.
// ---------------------------------------------------------------------------

typedef __attribute__((ext_vector_type(16))) __bf16 v16bf;
typedef __attribute__((ext_vector_type(8)))  __bf16 v8bf;
typedef __attribute__((ext_vector_type(8)))  float  v8f;
typedef __attribute__((ext_vector_type(4)))  float  v4f;

#define B_   16
#define L_   4096
#define D_   512
#define HD_  32
#define K_   512
#define CD_  64
#define ROWS_ (B_ * L_)            // 65536
#define TILE_ROWS 128
#define NBLK (ROWS_ / TILE_ROWS)   // 512

#define SCALE_PAR (0.25f / (65536.0f * 32.0f))
#define SCALE_SER (0.25f / (65536.0f * 64.0f))

__device__ __forceinline__ int laneId() { return threadIdx.x & 31; }
__device__ __forceinline__ int waveId() { return threadIdx.x >> 5; }

__device__ __forceinline__ v8f wmma_bf16(v16bf a, v16bf b, v8f c) {
  return __builtin_amdgcn_wmma_f32_16x16x32_bf16(false, a, false, b,
                                                 (short)0, c, false, false);
}
__device__ __forceinline__ unsigned short f2bf(float x) {
  __bf16 h = (__bf16)x;
  return __builtin_bit_cast(unsigned short, h);
}

// A-matrix 16x32 bf16 fragment (ISA 7.12.2 layout):
//  lanes 0-15 : M = lane,    K in {0..7} U {16..23}
//  lanes 16-31: M = lane-16, K in {8..15} U {24..31}
// rowPtr must already point at this lane's row (M = lane&15); 16B aligned.
__device__ __forceinline__ v16bf a_frag_f32(const float* rowPtr, int koff) {
  const int hf = (threadIdx.x >> 4) & 1;
  const float* p0 = rowPtr + koff + hf * 8;
  const float* p1 = rowPtr + koff + 16 + hf * 8;
  v4f x0 = *(const v4f*)p0;
  v4f x1 = *(const v4f*)(p0 + 4);
  v4f x2 = *(const v4f*)p1;
  v4f x3 = *(const v4f*)(p1 + 4);
  v16bf a;
#pragma unroll
  for (int j = 0; j < 4; ++j) {
    a[j]      = (__bf16)x0[j];
    a[4 + j]  = (__bf16)x1[j];
    a[8 + j]  = (__bf16)x2[j];
    a[12 + j] = (__bf16)x3[j];
  }
  return a;
}
__device__ __forceinline__ v16bf a_frag_bf(const unsigned short* rowPtr,
                                           int koff) {
  const int hf = (threadIdx.x >> 4) & 1;
  v8bf lo = *(const v8bf*)(rowPtr + koff + hf * 8);
  v8bf hi = *(const v8bf*)(rowPtr + koff + 16 + hf * 8);
  return __builtin_shufflevector(lo, hi, 0, 1, 2, 3, 4, 5, 6, 7, 8, 9, 10, 11,
                                 12, 13, 14, 15);
}

// B-matrix 32x16 bf16 fragment: B[k][n] = W[n][k], W pre-converted bf16:
//  lanes 0-15 : column N = nn0+lane,    K = kk0 + {0..7} U {16..23}
//  lanes 16-31: column N = nn0+lane-16, K = kk0 + {8..15} U {24..31}
// Pure 2x16B loads, zero VALU.
__device__ __forceinline__ v16bf b_frag_bf(const unsigned short* W, int ld,
                                           int nn0, int kk0) {
  const int n  = nn0 + (threadIdx.x & 15);
  const int kb = kk0 + ((threadIdx.x >> 4) & 1) * 8;
  const unsigned short* p = W + (size_t)n * ld + kb;
  v8bf lo = *(const v8bf*)p;
  v8bf hi = *(const v8bf*)(p + 16);
  return __builtin_shufflevector(lo, hi, 0, 1, 2, 3, 4, 5, 6, 7, 8, 9, 10, 11,
                                 12, 13, 14, 15);
}

// Euclidean-nearest-code VQ over a per-wave 16xDIMS tile staged in LDS.
// Distance GEMM via WMMA against 32 tiles of 16 codes; argmin carried as
// (|c|^2 - 2 z.c, idx) and reduced across the 16-lane half with shfl_xor.
template <int DIMS>
__device__ __forceinline__ void vq_stage(
    const float* stRow, int koff,            // lane's staged row, column offset
    const unsigned short* cbBf,              // bf16 codebook for WMMA B frags
    const float* cbF, const float* norms,    // f32 codebook (gather), |c|^2
    int rowBase, float* outIdx,              // indices written as float
    unsigned short* outE, int eld, int eoff, // gathered codes (bf16 buffer)
    float* totalLoss, float scale) {
  constexpr int NCH = DIMS / 32;
  const int m  = laneId() & 15;
  const int hf = laneId() >> 4;

  v16bf a[NCH];
#pragma unroll
  for (int c = 0; c < NCH; ++c) a[c] = a_frag_f32(stRow, koff + c * 32);

  float z2own = 0.f;  // |z|^2 of this lane's row (m)
#pragma unroll
  for (int j = 0; j < DIMS; j += 4) {
    v4f v = *(const v4f*)(stRow + koff + j);
    z2own += v[0] * v[0] + v[1] * v[1] + v[2] * v[2] + v[3] * v[3];
  }

  float minv[8];
  int   mini[8];
#pragma unroll
  for (int r = 0; r < 8; ++r) { minv[r] = 3.0e38f; mini[r] = 0; }

#pragma unroll 2
  for (int ct = 0; ct < 32; ++ct) {
    v8f d;
#pragma unroll
    for (int r = 0; r < 8; ++r) d[r] = 0.f;
#pragma unroll
    for (int c = 0; c < NCH; ++c)
      d = wmma_bf16(a[c], b_frag_bf(cbBf, DIMS, ct * 16, c * 32), d);
    const int code = ct * 16 + m;
    const float nrm = norms[code];
#pragma unroll
    for (int r = 0; r < 8; ++r) {
      float s = nrm - 2.0f * d[r];
      if (s < minv[r]) { minv[r] = s; mini[r] = code; }
    }
  }

#pragma unroll
  for (int r = 0; r < 8; ++r) {
#pragma unroll
    for (int mm = 1; mm < 16; mm <<= 1) {
      float ov = __shfl_xor(minv[r], mm, 32);
      int   oi = __shfl_xor(mini[r], mm, 32);
      if (ov < minv[r] || (ov == minv[r] && oi < mini[r])) {
        minv[r] = ov; mini[r] = oi;
      }
    }
  }

  float lossPart = 0.f;
#pragma unroll
  for (int r = 0; r < 8; ++r) {
    float z2r = __shfl(z2own, hf * 8 + r, 32);  // lane with m == hf*8+r
    lossPart += z2r + minv[r];                  // == |z - e|^2 for that row
  }
  if (m == 0) {
#pragma unroll
    for (int r = 0; r < 8; ++r)
      outIdx[rowBase + hf * 8 + r] = (float)mini[r];
    atomicAdd(totalLoss, lossPart * scale);
  }

  // gather e = cb[idx] into bf16 buffer; DIMS/16 dims per lane per row,
  // packed into one 4B (DPL=2) or 8B (DPL=4) store.
  constexpr int DPL = DIMS / 16;
#pragma unroll
  for (int r = 0; r < 8; ++r) {
    const int row = rowBase + hf * 8 + r;
    const float* crow = cbF + (size_t)mini[r] * DIMS;
    unsigned short* erow = outE + (size_t)row * eld + eoff;
    const int dc = m * DPL;
    if (DPL == 2) {
      unsigned int pk = (unsigned int)f2bf(crow[dc]) |
                        ((unsigned int)f2bf(crow[dc + 1]) << 16);
      *(unsigned int*)(erow + dc) = pk;
    } else {
      uint2 pk;
      pk.x = (unsigned int)f2bf(crow[dc]) |
             ((unsigned int)f2bf(crow[dc + 1]) << 16);
      pk.y = (unsigned int)f2bf(crow[dc + 2]) |
             ((unsigned int)f2bf(crow[dc + 3]) << 16);
      *(uint2*)(erow + dc) = pk;
    }
  }
}

// ---------------------------------------------------------------------------
__global__ __launch_bounds__(256, 1) void k_cvt(const float* __restrict__ s,
                                                unsigned short* __restrict__ d,
                                                int n) {
  const int i = blockIdx.x * 256 + threadIdx.x;
  if (i < n) d[i] = f2bf(s[i]);
}

__global__ __launch_bounds__(256, 1) void k_norms(
    const float* __restrict__ pcb, const float* __restrict__ cb0,
    const float* __restrict__ cb1, float* __restrict__ normp,
    float* __restrict__ n0, float* __restrict__ n1, float* __restrict__ total) {
  const int t = blockIdx.x * 256 + threadIdx.x;
  if (t == 0) *total = 0.f;
  if (t < 2 * K_) {
    float s = 0.f;
#pragma unroll
    for (int j = 0; j < HD_; ++j) { float v = pcb[(size_t)t * HD_ + j]; s += v * v; }
    normp[t] = s;
  } else if (t < 3 * K_) {
    const int c = t - 2 * K_;
    float s = 0.f;
#pragma unroll
    for (int j = 0; j < CD_; ++j) { float v = cb0[(size_t)c * CD_ + j]; s += v * v; }
    n0[c] = s;
  } else if (t < 4 * K_) {
    const int c = t - 3 * K_;
    float s = 0.f;
#pragma unroll
    for (int j = 0; j < CD_; ++j) { float v = cb1[(size_t)c * CD_ + j]; s += v * v; }
    n1[c] = s;
  }
}

// ---------------------------------------------------------------------------
__global__ __launch_bounds__(256, 1) void k_proj_vq(
    const float* __restrict__ x, const unsigned short* __restrict__ pwBf,
    const float* __restrict__ pb, const unsigned short* __restrict__ pcbBf,
    const float* __restrict__ pcb, const float* __restrict__ normp,
    unsigned short* __restrict__ cbuf, float* __restrict__ idxp,
    float* __restrict__ total) {
  __shared__ float stage[8][16][68];  // 34 KB, per-wave 16x64 (+4 pad)
  const int w  = waveId();
  const int m  = laneId() & 15;
  const int hf = laneId() >> 4;
  const int rowBase = blockIdx.x * TILE_ROWS + w * 16;
  const float* xrow = x + (size_t)(rowBase + m) * D_;

  v8f acc[4];
#pragma unroll
  for (int t = 0; t < 4; ++t)
#pragma unroll
    for (int r = 0; r < 8; ++r) acc[t][r] = 0.f;

  // z(16x64) = x(16x512) @ Wp^T, Wp flat (64,512) since pw is (2,32,512)
#pragma unroll 2
  for (int kc = 0; kc < 16; ++kc) {
    v16bf a = a_frag_f32(xrow, kc * 32);
#pragma unroll
    for (int nt = 0; nt < 4; ++nt)
      acc[nt] = wmma_bf16(a, b_frag_bf(pwBf, D_, nt * 16, kc * 32), acc[nt]);
  }
#pragma unroll
  for (int nt = 0; nt < 4; ++nt) {
    const int n = nt * 16 + m;
    const float bv = pb[n];
#pragma unroll
    for (int r = 0; r < 8; ++r) stage[w][hf * 8 + r][n] = acc[nt][r] + bv;
  }
  __builtin_amdgcn_wave_barrier();

  const float* stRow = &stage[w][m][0];
#pragma unroll
  for (int h = 0; h < 2; ++h)
    vq_stage<HD_>(stRow, h * HD_, pcbBf + (size_t)h * K_ * HD_,
                  pcb + (size_t)h * K_ * HD_, normp + h * K_, rowBase,
                  idxp + (size_t)h * ROWS_, cbuf, CD_, h * HD_, total,
                  SCALE_PAR);
}

// ---------------------------------------------------------------------------
__global__ __launch_bounds__(256, 1) void k_ref0(
    const unsigned short* __restrict__ cbuf,
    const unsigned short* __restrict__ w1, const float* __restrict__ b1,
    const unsigned short* __restrict__ w2, const float* __restrict__ b2,
    const unsigned short* __restrict__ cbBf, const float* __restrict__ cbF,
    const float* __restrict__ norms, unsigned short* __restrict__ e0buf,
    float* __restrict__ idx0, float* __restrict__ total) {
  __shared__ float stage[8][16][68];
  const int w  = waveId();
  const int m  = laneId() & 15;
  const int hf = laneId() >> 4;
  const int rowBase = blockIdx.x * TILE_ROWS + w * 16;
  const unsigned short* crow = cbuf + (size_t)(rowBase + m) * CD_;

  // hidden(16x32) = c(16x64) @ w1^T, w1 (32,64)
  v8f hid[2];
#pragma unroll
  for (int t = 0; t < 2; ++t)
#pragma unroll
    for (int r = 0; r < 8; ++r) hid[t][r] = 0.f;
#pragma unroll
  for (int kc = 0; kc < 2; ++kc) {
    v16bf a = a_frag_bf(crow, kc * 32);
#pragma unroll
    for (int nt = 0; nt < 2; ++nt)
      hid[nt] = wmma_bf16(a, b_frag_bf(w1, CD_, nt * 16, kc * 32), hid[nt]);
  }
#pragma unroll
  for (int nt = 0; nt < 2; ++nt) {
    const int n = nt * 16 + m;
    const float bv = b1[n];
#pragma unroll
    for (int r = 0; r < 8; ++r) {
      float v = hid[nt][r] + bv;  // exact GELU
      stage[w][hf * 8 + r][n] = 0.5f * v * (1.0f + erff(v * 0.70710678118f));
    }
  }
  __builtin_amdgcn_wave_barrier();

  // u0(16x64) = hidden(16x32) @ w2^T, w2 (64,32)
  const float* stRow = &stage[w][m][0];
  v16bf ah = a_frag_f32(stRow, 0);
  v8f u[4];
#pragma unroll
  for (int t = 0; t < 4; ++t)
#pragma unroll
    for (int r = 0; r < 8; ++r) u[t][r] = 0.f;
#pragma unroll
  for (int nt = 0; nt < 4; ++nt)
    u[nt] = wmma_bf16(ah, b_frag_bf(w2, HD_, nt * 16, 0), u[nt]);
  __builtin_amdgcn_wave_barrier();
#pragma unroll
  for (int nt = 0; nt < 4; ++nt) {
    const int n = nt * 16 + m;
    const float bv = b2[n];
#pragma unroll
    for (int r = 0; r < 8; ++r) stage[w][hf * 8 + r][n] = u[nt][r] + bv;
  }
  __builtin_amdgcn_wave_barrier();

  vq_stage<CD_>(stRow, 0, cbBf, cbF, norms, rowBase, idx0, e0buf, CD_, 0,
                total, SCALE_SER);
}

// ---------------------------------------------------------------------------
__global__ __launch_bounds__(256, 1) void k_ref1(
    const unsigned short* __restrict__ cbuf,
    const unsigned short* __restrict__ e0buf,
    const unsigned short* __restrict__ w1, const float* __restrict__ b1,
    const unsigned short* __restrict__ w2, const float* __restrict__ b2,
    const unsigned short* __restrict__ cbBf, const float* __restrict__ cbF,
    const float* __restrict__ norms, unsigned short* __restrict__ e1buf,
    float* __restrict__ idx1, float* __restrict__ total) {
  __shared__ float stage[8][16][68];
  const int w  = waveId();
  const int m  = laneId() & 15;
  const int hf = laneId() >> 4;
  const int rowBase = blockIdx.x * TILE_ROWS + w * 16;
  const unsigned short* crow = cbuf + (size_t)(rowBase + m) * CD_;
  const unsigned short* erow = e0buf + (size_t)(rowBase + m) * CD_;

  // hidden(16x64) = [c,e0](16x128) @ w1^T, w1 (64,128)
  v8f hid[4];
#pragma unroll
  for (int t = 0; t < 4; ++t)
#pragma unroll
    for (int r = 0; r < 8; ++r) hid[t][r] = 0.f;
#pragma unroll
  for (int kc = 0; kc < 4; ++kc) {
    v16bf a = (kc < 2) ? a_frag_bf(crow, kc * 32)
                       : a_frag_bf(erow, (kc - 2) * 32);
#pragma unroll
    for (int nt = 0; nt < 4; ++nt)
      hid[nt] = wmma_bf16(a, b_frag_bf(w1, 2 * CD_, nt * 16, kc * 32), hid[nt]);
  }
#pragma unroll
  for (int nt = 0; nt < 4; ++nt) {
    const int n = nt * 16 + m;
    const float bv = b1[n];
#pragma unroll
    for (int r = 0; r < 8; ++r) {
      float v = hid[nt][r] + bv;
      stage[w][hf * 8 + r][n] = 0.5f * v * (1.0f + erff(v * 0.70710678118f));
    }
  }
  __builtin_amdgcn_wave_barrier();

  // u1(16x64) = hidden(16x64) @ w2^T, w2 (64,64)
  const float* stRow = &stage[w][m][0];
  v16bf ah0 = a_frag_f32(stRow, 0);
  v16bf ah1 = a_frag_f32(stRow, 32);
  v8f u[4];
#pragma unroll
  for (int t = 0; t < 4; ++t)
#pragma unroll
    for (int r = 0; r < 8; ++r) u[t][r] = 0.f;
#pragma unroll
  for (int nt = 0; nt < 4; ++nt) {
    u[nt] = wmma_bf16(ah0, b_frag_bf(w2, CD_, nt * 16, 0), u[nt]);
    u[nt] = wmma_bf16(ah1, b_frag_bf(w2, CD_, nt * 16, 32), u[nt]);
  }
  __builtin_amdgcn_wave_barrier();
#pragma unroll
  for (int nt = 0; nt < 4; ++nt) {
    const int n = nt * 16 + m;
    const float bv = b2[n];
#pragma unroll
    for (int r = 0; r < 8; ++r) stage[w][hf * 8 + r][n] = u[nt][r] + bv;
  }
  __builtin_amdgcn_wave_barrier();

  vq_stage<CD_>(stRow, 0, cbBf, cbF, norms, rowBase, idx1, e1buf, CD_, 0,
                total, SCALE_SER);
}

// ---------------------------------------------------------------------------
__global__ __launch_bounds__(256, 1) void k_out(
    const unsigned short* __restrict__ cbuf,
    const unsigned short* __restrict__ e0buf,
    const unsigned short* __restrict__ e1buf,
    const unsigned short* __restrict__ outwBf, const float* __restrict__ outb,
    float* __restrict__ zq) {
  const int m  = laneId() & 15;
  const int hf = laneId() >> 4;
  const int rowBase = blockIdx.x * TILE_ROWS + waveId() * 16;
  const unsigned short* crow  = cbuf + (size_t)(rowBase + m) * CD_;
  const unsigned short* e0row = e0buf + (size_t)(rowBase + m) * CD_;
  const unsigned short* e1row = e1buf + (size_t)(rowBase + m) * CD_;

  // A fragments for final = [c, e0, e1] (16 x 192), resident across all tiles
  v16bf a[6];
  a[0] = a_frag_bf(crow, 0);  a[1] = a_frag_bf(crow, 32);
  a[2] = a_frag_bf(e0row, 0); a[3] = a_frag_bf(e0row, 32);
  a[4] = a_frag_bf(e1row, 0); a[5] = a_frag_bf(e1row, 32);

#pragma unroll 2
  for (int nt = 0; nt < 32; ++nt) {
    if (nt + 1 < 32)  // stream next out_w tile (global_prefetch_b8)
      __builtin_prefetch(outwBf + (size_t)((nt + 1) * 16 + m) * (3 * CD_), 0, 0);
    v8f acc;
#pragma unroll
    for (int r = 0; r < 8; ++r) acc[r] = 0.f;
#pragma unroll
    for (int c = 0; c < 6; ++c)
      acc = wmma_bf16(a[c], b_frag_bf(outwBf, 3 * CD_, nt * 16, c * 32), acc);
    const int n = nt * 16 + m;
    const float bv = outb[n];
#pragma unroll
    for (int r = 0; r < 8; ++r)
      zq[(size_t)(rowBase + hf * 8 + r) * D_ + n] = acc[r] + bv;
  }
}

// ---------------------------------------------------------------------------
extern "C" void kernel_launch(void* const* d_in, const int* in_sizes, int n_in,
                              void* d_out, int out_size, void* d_ws,
                              size_t ws_size, hipStream_t stream) {
  (void)in_sizes; (void)n_in; (void)out_size; (void)ws_size;
  const float* x    = (const float*)d_in[0];
  const float* pw   = (const float*)d_in[1];
  const float* pb   = (const float*)d_in[2];
  const float* pcb  = (const float*)d_in[3];
  const float* s0w1 = (const float*)d_in[4];
  const float* s0b1 = (const float*)d_in[5];
  const float* s0w2 = (const float*)d_in[6];
  const float* s0b2 = (const float*)d_in[7];
  const float* cbs0 = (const float*)d_in[8];
  const float* s1w1 = (const float*)d_in[9];
  const float* s1b1 = (const float*)d_in[10];
  const float* s1w2 = (const float*)d_in[11];
  const float* s1b2 = (const float*)d_in[12];
  const float* cbs1 = (const float*)d_in[13];
  const float* outw = (const float*)d_in[14];
  const float* outb = (const float*)d_in[15];

  // workspace layout (all segments 2KB-aligned)
  char* ws = (char*)d_ws;
  const size_t EB = (size_t)ROWS_ * CD_ * 2;  // 8 MiB
  unsigned short* cbuf = (unsigned short*)ws;
  unsigned short* e0b  = (unsigned short*)(ws + EB);
  unsigned short* e1b  = (unsigned short*)(ws + 2 * EB);
  char* p = ws + 3 * EB;
  float* normp = (float*)p;            p += 2 * K_ * 4;
  float* n0    = (float*)p;            p += K_ * 4;
  float* n1    = (float*)p;            p += K_ * 4;
  unsigned short* pwBf   = (unsigned short*)p; p += (size_t)CD_ * D_ * 2;
  unsigned short* pcbBf  = (unsigned short*)p; p += (size_t)2 * K_ * HD_ * 2;
  unsigned short* s0w1Bf = (unsigned short*)p; p += (size_t)HD_ * CD_ * 2;
  unsigned short* s0w2Bf = (unsigned short*)p; p += (size_t)CD_ * HD_ * 2;
  unsigned short* cb0Bf  = (unsigned short*)p; p += (size_t)K_ * CD_ * 2;
  unsigned short* s1w1Bf = (unsigned short*)p; p += (size_t)CD_ * 2 * CD_ * 2;
  unsigned short* s1w2Bf = (unsigned short*)p; p += (size_t)CD_ * CD_ * 2;
  unsigned short* cb1Bf  = (unsigned short*)p; p += (size_t)K_ * CD_ * 2;
  unsigned short* outwBf = (unsigned short*)p; p += (size_t)D_ * 3 * CD_ * 2;

  // d_out: z_q (ROWS_*D_) | idx_h0 | idx_h1 | idx0 | idx1 | total
  float* out = (float*)d_out;
  const size_t Z = (size_t)ROWS_ * D_;
  float* idxp  = out + Z;
  float* idx0  = out + Z + 2 * (size_t)ROWS_;
  float* idx1  = out + Z + 3 * (size_t)ROWS_;
  float* total = out + Z + 4 * (size_t)ROWS_;

  dim3 blk(256), grd(NBLK);
  auto cvt = [&](const float* s, unsigned short* d, int n) {
    k_cvt<<<dim3((n + 255) / 256), blk, 0, stream>>>(s, d, n);
  };
  cvt(pw,   pwBf,   CD_ * D_);
  cvt(pcb,  pcbBf,  2 * K_ * HD_);
  cvt(s0w1, s0w1Bf, HD_ * CD_);
  cvt(s0w2, s0w2Bf, CD_ * HD_);
  cvt(cbs0, cb0Bf,  K_ * CD_);
  cvt(s1w1, s1w1Bf, CD_ * 2 * CD_);
  cvt(s1w2, s1w2Bf, CD_ * CD_);
  cvt(cbs1, cb1Bf,  K_ * CD_);
  cvt(outw, outwBf, D_ * 3 * CD_);
  k_norms<<<dim3(8), blk, 0, stream>>>(pcb, cbs0, cbs1, normp, n0, n1, total);

  k_proj_vq<<<grd, blk, 0, stream>>>(x, pwBf, pb, pcbBf, pcb, normp, cbuf,
                                     idxp, total);
  k_ref0<<<grd, blk, 0, stream>>>(cbuf, s0w1Bf, s0b1, s0w2Bf, s0b2, cb0Bf,
                                  cbs0, n0, e0b, idx0, total);
  k_ref1<<<grd, blk, 0, stream>>>(cbuf, e0b, s1w1Bf, s1b1, s1w2Bf, s1b2,
                                  cb1Bf, cbs1, n1, e1b, idx1, total);
  k_out<<<grd, blk, 0, stream>>>(cbuf, e0b, e1b, outwBf, outb, out);
}
